// SimplifiedIFEBranch_31860067401864
// MI455X (gfx1250) — compile-verified, matrix operand order checked by reading, not run on hardware
//
#include <hip/hip_runtime.h>
#include <hip/hip_bf16.h>
#include <math.h>

// ---------------------------------------------------------------------------
// Types for CDNA5 WMMA (wave32, v_wmma_f32_16x16x32_f16)
// ---------------------------------------------------------------------------
typedef __attribute__((ext_vector_type(16))) _Float16 v16h;
typedef __attribute__((ext_vector_type(8)))  _Float16 v8h;
typedef __attribute__((ext_vector_type(8)))  float    v8f;

// Histogram constants (match reference exactly)
#define HIST_LO   (-3.2125f)         // -3.2 - (6.4/256)/2
#define HIST_HI   ( 3.1875f)         //  3.2 - (6.4/256)/2
#define HIST_INVW ( 5.0f)            // 1 / 0.2

// ---------------------------------------------------------------------------
// Kernel 1: per-image RGB-uv histogram -> f16 feature row [3072]
// One block per image, 256 threads (8 waves). 12KB LDS histogram + ds_add_f32.
// ---------------------------------------------------------------------------
__global__ void rgbuv_hist_kernel(const float* __restrict__ img,   // [32,3,512,512]
                                  _Float16* __restrict__ histH)    // [32,3072]
{
    __shared__ float h[3 * 1024];
    __shared__ float ssum[3];

    const int b = blockIdx.x;
    const int t = threadIdx.x;          // 0..255

    for (int p = t; p < 3 * 1024; p += 256) h[p] = 0.0f;
    if (t < 3) ssum[t] = 0.0f;
    __syncthreads();

    const float* base = img + (size_t)b * 3u * 512u * 512u;

    for (int p = t; p < 1024; p += 256) {
        const int y = p >> 5;           // 0..31
        const int x = p & 31;
        const int off = (y * 16) * 512 + (x * 16);   // nearest: floor(k*16)=16k
        const float R = base[off];
        const float G = base[512 * 512 + off];
        const float Bc = base[2 * 512 * 512 + off];

        const bool valid = (R > 0.0f) & (G > 0.0f) & (Bc > 0.0f);
        const float w = sqrtf(R * R + G * G + Bc * Bc);
        const float l0 = logf(R  > 0.0f ? R  : 1.0f);
        const float l1 = logf(G  > 0.0f ? G  : 1.0f);
        const float l2 = logf(Bc > 0.0f ? Bc : 1.0f);

        if (valid) {
            // i=0: Iu=l0-l2, Iv=l0-l1 ; i=1: Iu=l1-l2, Iv=l1-l0 ; i=2: Iu=l2-l1, Iv=l2-l0
            const float Ius[3] = { l0 - l2, l1 - l2, l2 - l1 };
            const float Ivs[3] = { l0 - l1, l1 - l0, l2 - l0 };
#pragma unroll
            for (int i = 0; i < 3; ++i) {
                const float Iu = Ius[i];
                const float Iv = Ivs[i];
                if (Iu >= HIST_LO && Iu <= HIST_HI && Iv >= HIST_LO && Iv <= HIST_HI) {
                    int iu = (int)floorf((Iu - HIST_LO) * HIST_INVW);
                    int iv = (int)floorf((Iv - HIST_LO) * HIST_INVW);
                    iu = iu < 0 ? 0 : (iu > 31 ? 31 : iu);
                    iv = iv < 0 ? 0 : (iv > 31 ? 31 : iv);
                    atomicAdd(&h[i * 1024 + iu * 32 + iv], w);   // ds_add_f32
                }
            }
        }
    }
    __syncthreads();

    // per-channel sums
#pragma unroll
    for (int i = 0; i < 3; ++i) {
        float partial = 0.0f;
        for (int p = t; p < 1024; p += 256) partial += h[i * 1024 + p];
        atomicAdd(&ssum[i], partial);
    }
    __syncthreads();

    _Float16* out = histH + (size_t)b * 3072u;
#pragma unroll
    for (int i = 0; i < 3; ++i) {
        const float inv = 1.0f / ssum[i];
        for (int p = t; p < 1024; p += 256)
            out[i * 1024 + p] = (_Float16)sqrtf(h[i * 1024 + p] * inv);
    }
}

// ---------------------------------------------------------------------------
// Kernel 2: f32 -> f16 weight conversion (one element per thread)
// ---------------------------------------------------------------------------
__global__ void cvt_f32_to_f16(const float* __restrict__ in,
                               _Float16* __restrict__ out, int n)
{
    const int i = blockIdx.x * blockDim.x + threadIdx.x;
    if (i < n) out[i] = (_Float16)in[i];
}

// ---------------------------------------------------------------------------
// Kernel 3: fused GEMM + bias + ReLU via v_wmma_f32_16x16x32_f16.
//   out[m,n] = relu( sum_k A[m,k] * W[n,k] + bias[n] ),  M = 32 (2 tiles)
// One wave computes one 16x16 tile. A and W share the same per-lane layout:
//   lane holds row (lane&15); K chunks {0..7,16..23} (lanes 0-15) or
//   {8..15,24..31} (lanes 16-31) -> two contiguous 16B loads per operand/step.
// ---------------------------------------------------------------------------
__device__ __forceinline__ v16h load_mat_row(const _Float16* __restrict__ base,
                                             int row, int K, int k, int hi)
{
    const _Float16* p = base + (size_t)row * (size_t)K + k + hi * 8;
    const v8h x0 = *reinterpret_cast<const v8h*>(p);        // K: k+hi*8 .. +7
    const v8h x1 = *reinterpret_cast<const v8h*>(p + 16);   // K: k+16+hi*8 .. +7
    v16h r;
#pragma unroll
    for (int i = 0; i < 8; ++i) { r[i] = x0[i]; r[i + 8] = x1[i]; }
    return r;
}

__global__ void gemm_bias_relu_wmma(const _Float16* __restrict__ A,   // [32,K]
                                    const _Float16* __restrict__ W,   // [N,K]
                                    const float* __restrict__ bias,   // [N]
                                    _Float16* __restrict__ outH,      // [32,N] or null
                                    float* __restrict__ outF,         // [32,N] or null
                                    int N, int K)
{
    const int wave = threadIdx.x >> 5;
    const int lane = threadIdx.x & 31;
    const int tile = blockIdx.x * (blockDim.x >> 5) + wave;   // grid sized exactly
    const int m0 = (tile & 1) * 16;
    const int n0 = (tile >> 1) * 16;
    const int r  = lane & 15;
    const int hi = lane >> 4;

    v8f acc = {};
    for (int k = 0; k < K; k += 32) {
        const v16h a = load_mat_row(A, m0 + r, K, k, hi);
        const v16h b = load_mat_row(W, n0 + r, K, k, hi);
        acc = __builtin_amdgcn_wmma_f32_16x16x32_f16(
            /*neg_a=*/false, a, /*neg_b=*/false, b,
            /*c_mod=*/(short)0, acc, /*reuse_a=*/false, /*reuse_b=*/false);
    }

    const float bn = bias[n0 + r];
#pragma unroll
    for (int v = 0; v < 8; ++v) {
        const int m = m0 + v + hi * 8;
        float val = acc[v] + bn;
        val = val > 0.0f ? val : 0.0f;
        const size_t idx = (size_t)m * (size_t)N + n0 + r;
        if (outF) outF[idx] = val;
        if (outH) outH[idx] = (_Float16)val;
    }
}

// ---------------------------------------------------------------------------
// Host side
// ---------------------------------------------------------------------------
extern "C" void kernel_launch(void* const* d_in, const int* in_sizes, int n_in,
                              void* d_out, int out_size, void* d_ws, size_t ws_size,
                              hipStream_t stream)
{
    (void)in_sizes; (void)n_in; (void)out_size; (void)ws_size;

    const float* img = (const float*)d_in[0];   // [32,3,512,512]
    const float* W1  = (const float*)d_in[1];   // [1024,3072]
    const float* b1  = (const float*)d_in[2];   // [1024]
    const float* W2  = (const float*)d_in[3];   // [512,1024]
    const float* b2  = (const float*)d_in[4];   // [512]
    const float* W3  = (const float*)d_in[5];   // [256,512]
    const float* b3  = (const float*)d_in[6];   // [256]
    float* out = (float*)d_out;                 // [32,256]

    // Workspace layout (f16 buffers, all 16B aligned)
    char* ws = (char*)d_ws;
    _Float16* histH = (_Float16*)(ws);                         // 32*3072*2 = 196608
    _Float16* W1h   = (_Float16*)(ws + 196608);                // 1024*3072*2 = 6291456
    _Float16* W2h   = (_Float16*)(ws + 196608 + 6291456);      // 512*1024*2  = 1048576
    _Float16* W3h   = (_Float16*)(ws + 196608 + 6291456 + 1048576);          // 262144
    _Float16* act1h = (_Float16*)(ws + 196608 + 6291456 + 1048576 + 262144); // 65536
    _Float16* act2h = (_Float16*)(ws + 196608 + 6291456 + 1048576 + 262144 + 65536); // 32768

    // 1) Histogram features (f16)
    rgbuv_hist_kernel<<<32, 256, 0, stream>>>(img, histH);

    // 2) Weight conversion f32 -> f16
    cvt_f32_to_f16<<<(1024 * 3072 + 255) / 256, 256, 0, stream>>>(W1, W1h, 1024 * 3072);
    cvt_f32_to_f16<<<(512 * 1024 + 255) / 256, 256, 0, stream>>>(W2, W2h, 512 * 1024);
    cvt_f32_to_f16<<<(256 * 512 + 255) / 256, 256, 0, stream>>>(W3, W3h, 256 * 512);

    // 3) MLP layers. tiles = 2*(N/16); 4 waves/block -> blocks = tiles/4 (exact).
    gemm_bias_relu_wmma<<<(2 * (1024 / 16)) / 4, 128, 0, stream>>>(
        histH, W1h, b1, act1h, nullptr, 1024, 3072);
    gemm_bias_relu_wmma<<<(2 * (512 / 16)) / 4, 128, 0, stream>>>(
        act1h, W2h, b2, act2h, nullptr, 512, 1024);
    gemm_bias_relu_wmma<<<(2 * (256 / 16)) / 4, 128, 0, stream>>>(
        act2h, W3h, b3, nullptr, out, 256, 512);
}
